// WindowAttenton_12610023981395
// MI455X (gfx1250) — compile-verified, matrix operand order checked by reading, not run on hardware
//
#include <hip/hip_runtime.h>
#include <hip/hip_bf16.h>
#include <cstdint>

typedef __attribute__((ext_vector_type(16))) __bf16 v16bf;
typedef __attribute__((ext_vector_type(8)))  float  v8f;
typedef __attribute__((ext_vector_type(4)))  __bf16 v4bf;
typedef int v4i __attribute__((vector_size(16)));  // matches builtin's pointee type

#define DIM_   512
#define NH_    16
#define HD_    32
#define NTOK   64      // tokens per window
#define NWIN   64      // windows per image
#define BW_    1024    // total windows
#define MROWS  (BW_ * NTOK)  // 65536 tokens
#define SCALE_F 0.17677669529663687f  // 32^-0.5

#define LDP 40  // padded LDS row stride in halfs (80B rows: conflict-free 16-lane reads)

// ---------------------------------------------------------------------------
// CDNA5 async global->LDS staging (guarded; falls back to VGPR bounce)
// ---------------------------------------------------------------------------
#if defined(__has_builtin)
#if __has_builtin(__builtin_amdgcn_global_load_async_to_lds_b128)
#define HAVE_ASYNC_LDS 1
#endif
#endif
#ifndef HAVE_ASYNC_LDS
#define HAVE_ASYNC_LDS 0
#endif

#if HAVE_ASYNC_LDS
#if __has_builtin(__builtin_amdgcn_s_wait_asynccnt)
#define WAIT_ASYNC(n) __builtin_amdgcn_s_wait_asynccnt(n)
#else
#define WAIT_ASYNC(n) asm volatile("s_wait_asynccnt %0" ::"i"(n) : "memory")
#endif
#else
#define WAIT_ASYNC(n) ((void)0)
#endif

// Copy 16B (8 bf16) global -> LDS. Async path uses ASYNCcnt-tracked DMA.
// AS1 ptr = full generic address; AS3 ptr = low 32 bits (LDS aperture layout).
__device__ __forceinline__ void copy_b128(const __bf16* g, __bf16* l) {
#if HAVE_ASYNC_LDS
  __builtin_amdgcn_global_load_async_to_lds_b128(
      (__attribute__((address_space(1))) v4i*)(uintptr_t)g,
      (__attribute__((address_space(3))) v4i*)(uint32_t)(uintptr_t)l, 0, 0);
#else
  *reinterpret_cast<int4*>(l) = *reinterpret_cast<const int4*>(g);
#endif
}

// ---------------------------------------------------------------------------
// WMMA helpers
// ---------------------------------------------------------------------------
__device__ __forceinline__ v8f wmma_bf16(v16bf a, v16bf b, v8f c) {
  return __builtin_amdgcn_wmma_f32_16x16x32_bf16(false, a, false, b, (short)0, c,
                                                 false, false);
}

// Per-lane 16-bf16 operand fragment from two 16B chunks (ISA 16-bit A layout:
// lanes 0-15 hold K=koff..koff+7 and K=16+koff.., koff = (lane>=16)*8).
__device__ __forceinline__ void load_frag(v16bf& f, const __bf16* p_lo, const __bf16* p_hi) {
  int4* fp = reinterpret_cast<int4*>(&f);
  fp[0] = *reinterpret_cast<const int4*>(p_lo);
  fp[1] = *reinterpret_cast<const int4*>(p_hi);
}

// ---------------------------------------------------------------------------
// Double-buffered GEMM mainloop: C(128x128) = A(MxK) * W(NxK)^T
// 256 threads = 8 waves, 2(M) x 4(N); each wave owns a 64x32 tile (4x2 WMMAs).
// CVT_A: A is fp32 and converted to bf16 while staging (saves a full
// conversion pass over x). NASYNC = async ops issued per iteration per wave.
// ---------------------------------------------------------------------------
template <int KDIM, bool CVT_A, int NASYNC>
__device__ __forceinline__ void gemm_tile(const void* __restrict__ Araw,
                                          const __bf16* __restrict__ W,
                                          int mt, int nt,
                                          __bf16 (*sA)[128][LDP],
                                          __bf16 (*sB)[128][LDP],
                                          v8f acc[4][2]) {
  const int tid  = threadIdx.x;
  const int lane = tid & 31;
  const int wave = tid >> 5;
  const int wm   = wave >> 2;        // 0..1
  const int wn   = wave & 3;         // 0..3
  const int lr   = lane & 15;
  const int koff = (lane >> 4) << 3; // 0 or 8

  const int ldr = tid >> 1;          // 0..127 staged row
  const int ldc = (tid & 1) << 4;    // 0/16 staged col base (16 halfs/thread)

  const float*  Af = (const float*)Araw;
  const __bf16* Ab = (const __bf16*)Araw;

  auto issue = [&](int k0, int b) {
    if (CVT_A) {
      const float* g = Af + (size_t)(mt * 128 + ldr) * KDIM + k0 + ldc;
#pragma unroll
      for (int c = 0; c < 4; ++c) {
        const float4 f = *reinterpret_cast<const float4*>(g + 4 * c);
        v4bf o;
        o[0] = (__bf16)f.x; o[1] = (__bf16)f.y; o[2] = (__bf16)f.z; o[3] = (__bf16)f.w;
        *reinterpret_cast<v4bf*>(&sA[b][ldr][ldc + 4 * c]) = o;
      }
    } else {
      const __bf16* g = Ab + (size_t)(mt * 128 + ldr) * KDIM + k0 + ldc;
      copy_b128(g, &sA[b][ldr][ldc]);
      copy_b128(g + 8, &sA[b][ldr][ldc + 8]);
    }
    const __bf16* gw = W + (size_t)(nt * 128 + ldr) * KDIM + k0 + ldc;
    copy_b128(gw, &sB[b][ldr][ldc]);
    copy_b128(gw + 8, &sB[b][ldr][ldc + 8]);
  };

  int buf = 0;
  issue(0, 0);  // prologue

  for (int k0 = 0; k0 < KDIM; k0 += 32) {
    const int nbuf = buf ^ 1;
    if (k0 + 32 < KDIM) {
      issue(k0 + 32, nbuf);      // overlap next-tile DMA with this tile's WMMAs
      WAIT_ASYNC(NASYNC);        // in-order: current buffer's loads are done
    } else {
      WAIT_ASYNC(0);
    }
    __syncthreads();             // all waves' writes to sX[buf] visible

    v16bf bf0, bf1;
    { const __bf16* s = &sB[buf][wn * 32 + lr][0];      load_frag(bf0, s + koff, s + 16 + koff); }
    { const __bf16* s = &sB[buf][wn * 32 + 16 + lr][0]; load_frag(bf1, s + koff, s + 16 + koff); }

#pragma unroll
    for (int mb = 0; mb < 4; ++mb) {
      v16bf af;
      const __bf16* s = &sA[buf][wm * 64 + mb * 16 + lr][0];
      load_frag(af, s + koff, s + 16 + koff);
      acc[mb][0] = wmma_bf16(af, bf0, acc[mb][0]);
      acc[mb][1] = wmma_bf16(af, bf1, acc[mb][1]);
    }
    __syncthreads();             // done reading sX[buf] before it is re-filled
    buf = nbuf;
  }
}

__device__ __forceinline__ void zero_acc(v8f acc[4][2]) {
#pragma unroll
  for (int i = 0; i < 4; ++i)
#pragma unroll
    for (int j = 0; j < 2; ++j)
#pragma unroll
      for (int e = 0; e < 8; ++e) acc[i][j][e] = 0.f;
}

// ---------------------------------------------------------------------------
// Kernel 1: QKV projection. Reads x fp32 directly (bf16 convert in staging).
// Scatters bf16 into q (pre-scaled), k, and v transposed (BW, NH, HD, N).
// ---------------------------------------------------------------------------
__global__ __launch_bounds__(256) void gemm_qkv_kernel(
    const float* __restrict__ x, const __bf16* __restrict__ wqkv,
    const float* __restrict__ qkv_b,
    __bf16* __restrict__ qo, __bf16* __restrict__ ko, __bf16* __restrict__ vto) {
  __shared__ __align__(16) __bf16 sA[2][128][LDP];
  __shared__ __align__(16) __bf16 sB[2][128][LDP];
  const int MT = MROWS / 128;       // 512
  const int mt = blockIdx.x % MT;
  const int nt = blockIdx.x / MT;   // 0..11

  v8f acc[4][2];
  zero_acc(acc);
  gemm_tile<DIM_, true, 2>(x, wqkv, mt, nt, sA, sB, acc);

  const int lane = threadIdx.x & 31;
  const int wave = threadIdx.x >> 5;
  const int wm = wave >> 2, wn = wave & 3;
  const int lr = lane & 15, lh = lane >> 4;

#pragma unroll
  for (int mb = 0; mb < 4; ++mb)
#pragma unroll
    for (int nb = 0; nb < 2; ++nb)
#pragma unroll
      for (int e = 0; e < 8; ++e) {
        const int m = mt * 128 + wm * 64 + mb * 16 + e + lh * 8;
        const int n = nt * 128 + wn * 32 + nb * 16 + lr;
        float v = acc[mb][nb][e] + qkv_b[n];
        const int which = n >> 9;          // 0=q 1=k 2=v
        const int hh = (n >> 5) & 15;      // head
        const int dd = n & 31;             // head dim
        const int bw = m >> 6, t = m & 63; // window, token
        const size_t hb = (size_t)bw * NH_ + hh;
        if (which == 0)      qo[(hb * NTOK + t) * HD_ + dd]  = (__bf16)(v * SCALE_F);
        else if (which == 1) ko[(hb * NTOK + t) * HD_ + dd]  = (__bf16)v;
        else                 vto[(hb * HD_ + dd) * NTOK + t] = (__bf16)v;
      }
}

// ---------------------------------------------------------------------------
// Kernel 2: per-(window, head) attention. 4 waves/block, 1 head per wave.
// ---------------------------------------------------------------------------
__global__ __launch_bounds__(128) void attn_kernel(
    const __bf16* __restrict__ qb, const __bf16* __restrict__ kb,
    const __bf16* __restrict__ vtb, const float* __restrict__ bias_g,
    const float* __restrict__ mask, __bf16* __restrict__ yb) {
  __shared__ __align__(16) __bf16 sP[4][NTOK][NTOK];  // 32 KB per-wave P tiles

  const int wave = threadIdx.x >> 5;
  const int lane = threadIdx.x & 31;
  const int lr = lane & 15, lh = lane >> 4;
  const int koff = lh << 3;

  const int bw = blockIdx.x >> 2;
  const int h  = ((blockIdx.x & 3) << 2) + wave;
  const int w  = bw & (NWIN - 1);

  const size_t hb = (size_t)bw * NH_ + h;
  const __bf16* qh = qb  + hb * (NTOK * HD_);
  const __bf16* kh = kb  + hb * (NTOK * HD_);
  const __bf16* vh = vtb + hb * (HD_ * NTOK);

  // ---- S = q * k^T (K = HD = 32: one WMMA per 16x16 tile) ----
  v8f s[4][4];
#pragma unroll
  for (int i = 0; i < 4; ++i)
#pragma unroll
    for (int j = 0; j < 4; ++j)
#pragma unroll
      for (int e = 0; e < 8; ++e) s[i][j][e] = 0.f;

  v16bf qf[4];
#pragma unroll
  for (int mb = 0; mb < 4; ++mb) {
    const __bf16* p = qh + (size_t)(mb * 16 + lr) * HD_;
    load_frag(qf[mb], p + koff, p + 16 + koff);
  }
#pragma unroll
  for (int nb = 0; nb < 4; ++nb) {
    v16bf kf;
    const __bf16* p = kh + (size_t)(nb * 16 + lr) * HD_;
    load_frag(kf, p + koff, p + 16 + koff);
#pragma unroll
    for (int mb = 0; mb < 4; ++mb) s[mb][nb] = wmma_bf16(qf[mb], kf, s[mb][nb]);
  }

  // ---- +bias +mask (C layout: VGPR e -> row e+8*lh, lane%16 -> col) ----
  const float* bh = bias_g + (size_t)h * (NTOK * NTOK);
  const float* mh = mask + (size_t)w * (NTOK * NTOK);
#pragma unroll
  for (int mb = 0; mb < 4; ++mb)
#pragma unroll
    for (int e = 0; e < 8; ++e) {
      const int m = mb * 16 + e + lh * 8;
#pragma unroll
      for (int nb = 0; nb < 4; ++nb) {
        const int n = nb * 16 + lr;
        s[mb][nb][e] += bh[m * NTOK + n] + mh[m * NTOK + n];
      }
    }

  // ---- row softmax (row lives in one VGPR across a 16-lane half) ----
#pragma unroll
  for (int mb = 0; mb < 4; ++mb)
#pragma unroll
    for (int e = 0; e < 8; ++e) {
      float mx = s[mb][0][e];
#pragma unroll
      for (int nb = 1; nb < 4; ++nb) mx = fmaxf(mx, s[mb][nb][e]);
#pragma unroll
      for (int off = 1; off < 16; off <<= 1) mx = fmaxf(mx, __shfl_xor(mx, off, 32));
      float sm = 0.f;
#pragma unroll
      for (int nb = 0; nb < 4; ++nb) {
        float p = __expf(s[mb][nb][e] - mx);
        s[mb][nb][e] = p;
        sm += p;
      }
#pragma unroll
      for (int off = 1; off < 16; off <<= 1) sm += __shfl_xor(sm, off, 32);
      const float inv = __frcp_rn(sm);
#pragma unroll
      for (int nb = 0; nb < 4; ++nb) s[mb][nb][e] *= inv;
    }

  // ---- P -> LDS bf16 (C-layout scatter), reload in A-operand layout ----
#pragma unroll
  for (int mb = 0; mb < 4; ++mb)
#pragma unroll
    for (int nb = 0; nb < 4; ++nb)
#pragma unroll
      for (int e = 0; e < 8; ++e)
        sP[wave][mb * 16 + e + lh * 8][nb * 16 + lr] = (__bf16)s[mb][nb][e];
  __syncthreads();

  // ---- O = P * V : K = 64 (2 chunks), N = 32 (2 blocks) ----
  v16bf vf[2][2];
#pragma unroll
  for (int nb2 = 0; nb2 < 2; ++nb2)
#pragma unroll
    for (int kc = 0; kc < 2; ++kc) {
      const __bf16* p = vh + (size_t)(nb2 * 16 + lr) * NTOK + kc * 32;
      load_frag(vf[nb2][kc], p + koff, p + 16 + koff);
    }

  v8f o[4][2];
#pragma unroll
  for (int i = 0; i < 4; ++i)
#pragma unroll
    for (int j = 0; j < 2; ++j)
#pragma unroll
      for (int e = 0; e < 8; ++e) o[i][j][e] = 0.f;

#pragma unroll
  for (int mb = 0; mb < 4; ++mb)
#pragma unroll
    for (int kc = 0; kc < 2; ++kc) {
      v16bf pf;
      const __bf16* p = &sP[wave][mb * 16 + lr][kc * 32];
      load_frag(pf, p + koff, p + 16 + koff);
      o[mb][0] = wmma_bf16(pf, vf[0][kc], o[mb][0]);
      o[mb][1] = wmma_bf16(pf, vf[1][kc], o[mb][1]);
    }

#pragma unroll
  for (int mb = 0; mb < 4; ++mb)
#pragma unroll
    for (int nb2 = 0; nb2 < 2; ++nb2)
#pragma unroll
      for (int e = 0; e < 8; ++e) {
        const int m = mb * 16 + e + lh * 8;
        const int col = h * HD_ + nb2 * 16 + lr;
        yb[((size_t)bw * NTOK + m) * DIM_ + col] = (__bf16)o[mb][nb2][e];
      }
}

// ---------------------------------------------------------------------------
// Kernel 3: output projection -> fp32 d_out (both operands bf16: full async)
// ---------------------------------------------------------------------------
__global__ __launch_bounds__(256) void gemm_proj_kernel(
    const __bf16* __restrict__ yb, const __bf16* __restrict__ wproj,
    const float* __restrict__ proj_b, float* __restrict__ out) {
  __shared__ __align__(16) __bf16 sA[2][128][LDP];
  __shared__ __align__(16) __bf16 sB[2][128][LDP];
  const int MT = MROWS / 128;
  const int mt = blockIdx.x % MT;
  const int nt = blockIdx.x / MT;  // 0..3

  v8f acc[4][2];
  zero_acc(acc);
  gemm_tile<DIM_, false, 4>(yb, wproj, mt, nt, sA, sB, acc);

  const int lane = threadIdx.x & 31;
  const int wave = threadIdx.x >> 5;
  const int wm = wave >> 2, wn = wave & 3;
  const int lr = lane & 15, lh = lane >> 4;

#pragma unroll
  for (int mb = 0; mb < 4; ++mb)
#pragma unroll
    for (int nb = 0; nb < 2; ++nb)
#pragma unroll
      for (int e = 0; e < 8; ++e) {
        const int m = mt * 128 + wm * 64 + mb * 16 + e + lh * 8;
        const int n = nt * 128 + wn * 32 + nb * 16 + lr;
        out[(size_t)m * DIM_ + n] = acc[mb][nb][e] + proj_b[n];
      }
}

// ---------------------------------------------------------------------------
// Elementwise f32 -> bf16 (weights only)
// ---------------------------------------------------------------------------
__global__ void cvt_bf16_kernel(const float* __restrict__ in, __bf16* __restrict__ out, int n) {
  const int i = (blockIdx.x * blockDim.x + threadIdx.x) << 2;
  if (i < n) {
    const float4 v = *reinterpret_cast<const float4*>(in + i);
    v4bf o;
    o[0] = (__bf16)v.x; o[1] = (__bf16)v.y; o[2] = (__bf16)v.z; o[3] = (__bf16)v.w;
    *reinterpret_cast<v4bf*>(out + i) = o;
  }
}

// Gather relative-position bias into dense (NH, N, N) fp32
__global__ void build_bias_kernel(const float* __restrict__ table,
                                  const int* __restrict__ rel_index,
                                  float* __restrict__ bias_g) {
  const int i = blockIdx.x * blockDim.x + threadIdx.x;  // NH*N*N = 65536
  const int h = i >> 12;
  const int mn = i & 4095;
  bias_g[i] = table[rel_index[mn] * NH_ + h];
}

// ---------------------------------------------------------------------------
// Host launcher
// ---------------------------------------------------------------------------
extern "C" void kernel_launch(void* const* d_in, const int* in_sizes, int n_in,
                              void* d_out, int out_size, void* d_ws, size_t ws_size,
                              hipStream_t stream) {
  (void)in_sizes; (void)n_in; (void)out_size; (void)ws_size;
  const float* x      = (const float*)d_in[0];
  const float* mask   = (const float*)d_in[1];
  const float* qkv_w  = (const float*)d_in[2];
  const float* qkv_b  = (const float*)d_in[3];
  const float* table  = (const float*)d_in[4];
  const float* proj_w = (const float*)d_in[5];
  const float* proj_b = (const float*)d_in[6];
  const int*   relidx = (const int*)d_in[7];
  float* out = (float*)d_out;

  char* ws = (char*)d_ws;
  size_t off = 0;
  auto alloc = [&](size_t bytes) -> char* {
    char* p = ws + off;
    off += (bytes + 255) & ~(size_t)255;
    return p;
  };
  const size_t act = (size_t)MROWS * DIM_;  // 33.5M elements
  __bf16* qb     = (__bf16*)alloc(act * 2);
  __bf16* kb     = (__bf16*)alloc(act * 2);
  __bf16* vtb    = (__bf16*)alloc(act * 2);
  __bf16* yb     = (__bf16*)alloc(act * 2);
  __bf16* wqkvb  = (__bf16*)alloc((size_t)3 * DIM_ * DIM_ * 2);
  __bf16* wprojb = (__bf16*)alloc((size_t)DIM_ * DIM_ * 2);
  float*  biasg  = (float*)alloc((size_t)NH_ * NTOK * NTOK * 4);

  cvt_bf16_kernel<<<3 * DIM_ * DIM_ / 1024, 256, 0, stream>>>(qkv_w, wqkvb, 3 * DIM_ * DIM_);
  cvt_bf16_kernel<<<DIM_ * DIM_ / 1024, 256, 0, stream>>>(proj_w, wprojb, DIM_ * DIM_);
  build_bias_kernel<<<NH_ * NTOK * NTOK / 256, 256, 0, stream>>>(table, relidx, biasg);

  gemm_qkv_kernel<<<(MROWS / 128) * (3 * DIM_ / 128), 256, 0, stream>>>(
      x, wqkvb, qkv_b, qb, kb, vtb);

  attn_kernel<<<BW_ * 4, 128, 0, stream>>>(qb, kb, vtb, biasg, mask, yb);

  gemm_proj_kernel<<<(MROWS / 128) * (DIM_ / 128), 256, 0, stream>>>(
      yb, wprojb, proj_b, out);
}
